// GNN_46059229282625
// MI455X (gfx1250) — matrix operand order repaired and checked by reading, not compiled, not run on hardware
//
#include <hip/hip_runtime.h>
#include <hip/hip_bf16.h>

// ---------------------------------------------------------------------------
// GIN forward for MI455X (gfx1250): wave32, WMMA f32<=f16 16x16x32
// ---------------------------------------------------------------------------

typedef __attribute__((ext_vector_type(16))) _Float16 v16h;
typedef __attribute__((ext_vector_type(8)))  _Float16 v8h;
typedef __attribute__((ext_vector_type(8)))  float    v8f;

#define D_EMB 128
#define BN_EPS 1e-5f

// ---------------------------------------------------------------------------
// Hardware FP32 atomic add (global_atomic_add_f32, no CAS loop)
// ---------------------------------------------------------------------------
__device__ __forceinline__ void atomic_add_f32(float* p, float v) {
    unsafeAtomicAdd(p, v);
}

// ---------------------------------------------------------------------------
// h[i,:] = atom_emb1[x[i,0]] + atom_emb2[x[i,1]]   (float4 per thread)
// ---------------------------------------------------------------------------
__global__ void atom_encode_kernel(const int* __restrict__ x,
                                   const float* __restrict__ emb1,
                                   const float* __restrict__ emb2,
                                   float* __restrict__ h, int N) {
    int idx = blockIdx.x * blockDim.x + threadIdx.x;       // N*32 threads
    if (idx >= N * 32) return;
    int i  = idx >> 5;
    int d4 = (idx & 31) << 2;
    int t0 = x[2 * i + 0];
    int t1 = x[2 * i + 1];
    float4 a = *(const float4*)(emb1 + (size_t)t0 * D_EMB + d4);
    float4 b = *(const float4*)(emb2 + (size_t)t1 * D_EMB + d4);
    float4 o; o.x = a.x + b.x; o.y = a.y + b.y; o.z = a.z + b.z; o.w = a.w + b.w;
    *(float4*)(h + (size_t)i * D_EMB + d4) = o;
}

// ---------------------------------------------------------------------------
// zero a float buffer (float4 per thread)
// ---------------------------------------------------------------------------
__global__ void zero_f32_kernel(float* __restrict__ p, int n4) {
    int idx = blockIdx.x * blockDim.x + threadIdx.x;
    if (idx >= n4) return;
    float4 z = {0.f, 0.f, 0.f, 0.f};
    *(float4*)(p + (size_t)idx * 4) = z;
}

// ---------------------------------------------------------------------------
// Convert this layer's weights to f16, transposed to [n][k] so that a WMMA
// B-fragment is a single contiguous 32B load per lane. Also zeroes the BN
// accumulators (first 128 threads).
// ---------------------------------------------------------------------------
__global__ void convert_weights_kernel(const float* __restrict__ W1, // [128][256]
                                       const float* __restrict__ W2, // [256][128]
                                       _Float16* __restrict__ w1t,   // [256][128]
                                       _Float16* __restrict__ w2t,   // [128][256]
                                       float* __restrict__ colsum,
                                       float* __restrict__ colsumsq) {
    int idx = blockIdx.x * blockDim.x + threadIdx.x;       // 65536 threads
    if (idx < D_EMB) { colsum[idx] = 0.f; colsumsq[idx] = 0.f; }
    if (idx < 128 * 256) {
        int k = idx >> 8, n = idx & 255;                   // W1[k][n]
        w1t[(size_t)n * 128 + k] = (_Float16)W1[idx];
    } else if (idx < 2 * 128 * 256) {
        int j = idx - 128 * 256;
        int k = j >> 7, n = j & 127;                       // W2[k][n]
        w2t[(size_t)n * 256 + k] = (_Float16)W2[j];
    }
}

// ---------------------------------------------------------------------------
// Scatter: one wave per (edge | self-loop); lane handles 4 of 128 features.
//   agg[dst] += h[src] + bond_emb1[ea0] + bond_emb2[ea1]
// Self-loop e >= E: src=dst=e-E, ea0=6, ea1=3.
// ---------------------------------------------------------------------------
__global__ void scatter_kernel(const float* __restrict__ h,
                               const int* __restrict__ edge_index, // [2*E]
                               const int* __restrict__ edge_attr,  // [E*2]
                               const float* __restrict__ be1,      // [7][128]
                               const float* __restrict__ be2,      // [4][128]
                               float* __restrict__ agg,
                               int E, int N) {
    int gid  = blockIdx.x * blockDim.x + threadIdx.x;
    int e    = gid >> 5;
    int lane = threadIdx.x & 31;
    int total = E + N;
    if (e >= total) return;
    int src, dst, a0, a1;
    if (e < E) {
        src = edge_index[e];
        dst = edge_index[E + e];
        a0  = edge_attr[2 * e + 0];
        a1  = edge_attr[2 * e + 1];
    } else {
        src = dst = e - E; a0 = 6; a1 = 3;
    }
    int d = lane << 2;
    float4 hv = *(const float4*)(h   + (size_t)src * D_EMB + d);
    float4 e1 = *(const float4*)(be1 + (size_t)a0  * D_EMB + d);
    float4 e2 = *(const float4*)(be2 + (size_t)a1  * D_EMB + d);
    float* out = agg + (size_t)dst * D_EMB + d;
    atomic_add_f32(out + 0, hv.x + e1.x + e2.x);
    atomic_add_f32(out + 1, hv.y + e1.y + e2.y);
    atomic_add_f32(out + 2, hv.z + e1.z + e2.z);
    atomic_add_f32(out + 3, hv.w + e1.w + e2.w);
}

// ---------------------------------------------------------------------------
// Fused MLP:  tmp = relu(agg @ W1 + b1) @ W2 + b2,  plus BN partial sums.
// 128 threads = 4 waves; each wave owns a 16-row tile (64 rows per block).
// GEMM1: K=128 (4 k-steps), Ncols=256 (16 col-tiles)
// GEMM2: K=256 (8 k-steps), Ncols=128 (8 col-tiles)
// Intermediate restaged via LDS to convert D-layout -> A-layout.
// ---------------------------------------------------------------------------
__launch_bounds__(128)
__global__ void mlp_kernel(const float* __restrict__ agg,      // [N_pad][128]
                           const _Float16* __restrict__ w1t,   // [256][128]
                           const _Float16* __restrict__ w2t,   // [128][256]
                           const float* __restrict__ b1,       // [256]
                           const float* __restrict__ b2,       // [128]
                           float* __restrict__ tmp,            // [N_pad][128]
                           float* __restrict__ colsum,         // [128]
                           float* __restrict__ colsumsq,       // [128]
                           int N) {
    __shared__ __align__(32) _Float16 t_sh[4][16 * 256];       // 32 KB

    const int lane = threadIdx.x & 31;
    const int wave = threadIdx.x >> 5;
    const int tile = blockIdx.x * 4 + wave;                    // 16-row tile
    const int row0 = tile * 16;
    const int m  = lane & 15;                                  // row / col within frag
    const int hi = lane >> 4;

    // ---- A fragments for GEMM1: rows row0..row0+15, K = 0..127 (fp32 -> f16)
    // A frag, contiguous k order used here (matches B): lane holds row m; halfs j -> K = k0+16*hi+j
    v16h afr[4];
    {
        const float* ap = agg + (size_t)(row0 + m) * D_EMB + (hi << 4);
#pragma unroll
        for (int kk = 0; kk < 4; ++kk) {
            const float* p = ap + kk * 32;
            v16h a;
#pragma unroll
            for (int j = 0; j < 8; ++j) {
                a[j]     = (_Float16)p[j];
                a[8 + j] = (_Float16)p[8 + j];
            }
            afr[kk] = a;
        }
    }

    // ---- GEMM1 + bias + ReLU -> stage f16 into LDS (row-major [16][256])
    _Float16* ts = &t_sh[wave][0];
#pragma unroll
    for (int nt = 0; nt < 16; ++nt) {
        const int n0 = nt * 16;
        v8f c = {};
#pragma unroll
        for (int kk = 0; kk < 4; ++kk) {
            // B frag, contiguous k order used here (matches A): lane holds column m; halfs j -> K = 16*hi + j
            const int kb = kk * 32 + (hi << 4);
            v16h b = *(const v16h*)(w1t + (size_t)(n0 + m) * 128 + kb);
            c = __builtin_amdgcn_wmma_f32_16x16x32_f16(
                    false, afr[kk], false, b, (short)0, c, false, false);
        }
        const float bias = b1[n0 + m];
#pragma unroll
        for (int r = 0; r < 8; ++r) {
            float v = c[r] + bias;                   // D-layout: row r+8*hi, col m
            v = v > 0.f ? v : 0.f;
            ts[(r + (hi << 3)) * 256 + n0 + m] = (_Float16)v;
        }
    }
    __syncthreads();

    // ---- A fragments for GEMM2 from LDS (K = 0..255)
    v16h tfr[8];
#pragma unroll
    for (int kk = 0; kk < 8; ++kk) {
        const int ks = kk * 32 + (hi << 4);
        v8h lo  = *(const v8h*)(ts + m * 256 + ks);
        v8h hi8 = *(const v8h*)(ts + m * 256 + ks + 8);
        v16h a;
#pragma unroll
        for (int j = 0; j < 8; ++j) { a[j] = lo[j]; a[8 + j] = hi8[j]; }
        tfr[kk] = a;
    }

    // ---- GEMM2 + bias -> tmp, plus BN column sums / sums of squares
#pragma unroll
    for (int nt = 0; nt < 8; ++nt) {
        const int n0 = nt * 16;
        v8f c = {};
#pragma unroll
        for (int kk = 0; kk < 8; ++kk) {
            const int kb = kk * 32 + (hi << 4);
            v16h b = *(const v16h*)(w2t + (size_t)(n0 + m) * 256 + kb);
            c = __builtin_amdgcn_wmma_f32_16x16x32_f16(
                    false, tfr[kk], false, b, (short)0, c, false, false);
        }
        const float bias = b2[n0 + m];
        float s = 0.f, ss = 0.f;
#pragma unroll
        for (int r = 0; r < 8; ++r) {
            float v = c[r] + bias;
            int row = row0 + r + (hi << 3);
            tmp[(size_t)row * D_EMB + n0 + m] = v;
            if (row < N) { s += v; ss += v * v; }    // mask padded rows from BN
        }
        atomic_add_f32(&colsum[n0 + m], s);
        atomic_add_f32(&colsumsq[n0 + m], ss);
    }
}

// ---------------------------------------------------------------------------
// BatchNorm finalize (+ ReLU except last layer):  h = BN(tmp)
// ---------------------------------------------------------------------------
__global__ void bn_finalize_kernel(const float* __restrict__ tmp,
                                   const float* __restrict__ colsum,
                                   const float* __restrict__ colsumsq,
                                   const float* __restrict__ gamma,
                                   const float* __restrict__ beta,
                                   float* __restrict__ h,
                                   int N, int do_relu) {
    int idx = blockIdx.x * blockDim.x + threadIdx.x;
    if (idx >= N * D_EMB) return;
    int n = idx & (D_EMB - 1);
    float invN = 1.0f / (float)N;
    float mu   = colsum[n] * invN;
    float var  = colsumsq[n] * invN - mu * mu;
    float y = (tmp[idx] - mu) * rsqrtf(var + BN_EPS) * gamma[n] + beta[n];
    if (do_relu) y = fmaxf(y, 0.f);
    h[idx] = y;
}

// ---------------------------------------------------------------------------
// Host launcher
// ---------------------------------------------------------------------------
extern "C" void kernel_launch(void* const* d_in, const int* in_sizes, int n_in,
                              void* d_out, int out_size, void* d_ws, size_t ws_size,
                              hipStream_t stream) {
    const int*   x          = (const int*)d_in[0];
    const int*   edge_index = (const int*)d_in[1];
    const int*   edge_attr  = (const int*)d_in[2];
    const float* atom_emb1  = (const float*)d_in[3];
    const float* atom_emb2  = (const float*)d_in[4];
    const float* bond_emb1  = (const float*)d_in[5];   // [5][7][128]
    const float* bond_emb2  = (const float*)d_in[6];   // [5][4][128]
    const float* W1         = (const float*)d_in[7];   // [5][128][256]
    const float* b1         = (const float*)d_in[8];   // [5][256]
    const float* W2         = (const float*)d_in[9];   // [5][256][128]
    const float* b2         = (const float*)d_in[10];  // [5][128]
    const float* gamma      = (const float*)d_in[11];  // [5][128]
    const float* beta       = (const float*)d_in[12];  // [5][128]

    const int N = in_sizes[0] / 2;
    const int E = in_sizes[1] / 2;
    const int N_pad = (N + 63) & ~63;                  // multiple of 64 rows

    // workspace layout
    char* ws = (char*)d_ws;
    float*    agg      = (float*)ws;       ws += (size_t)N_pad * D_EMB * sizeof(float);
    float*    tmp      = (float*)ws;       ws += (size_t)N_pad * D_EMB * sizeof(float);
    _Float16* w1t      = (_Float16*)ws;    ws += (size_t)128 * 256 * sizeof(_Float16);
    _Float16* w2t      = (_Float16*)ws;    ws += (size_t)256 * 128 * sizeof(_Float16);
    float*    colsum   = (float*)ws;       ws += D_EMB * sizeof(float);
    float*    colsumsq = (float*)ws;       ws += D_EMB * sizeof(float);

    float* h = (float*)d_out;                          // node features, reused per layer

    // AtomEncoder
    {
        int threads = N * 32;
        atom_encode_kernel<<<(threads + 255) / 256, 256, 0, stream>>>(
            x, atom_emb1, atom_emb2, h, N);
    }

    for (int l = 0; l < 5; ++l) {
        // zero agg (padded) -- also zero-pads rows N..N_pad for clean WMMA tiles
        {
            int n4 = N_pad * (D_EMB / 4);
            zero_f32_kernel<<<(n4 + 255) / 256, 256, 0, stream>>>(agg, n4);
        }
        // convert this layer's weights + zero BN accumulators
        convert_weights_kernel<<<(2 * 128 * 256 + 255) / 256, 256, 0, stream>>>(
            W1 + (size_t)l * 128 * 256, W2 + (size_t)l * 256 * 128,
            w1t, w2t, colsum, colsumsq);
        // scatter-add messages (edges + self-loops)
        {
            int waves = E + N;
            int threads = waves * 32;
            scatter_kernel<<<(threads + 255) / 256, 256, 0, stream>>>(
                h, edge_index, edge_attr,
                bond_emb1 + (size_t)l * 7 * D_EMB,
                bond_emb2 + (size_t)l * 4 * D_EMB,
                agg, E, N);
        }
        // fused WMMA MLP + BN stats
        mlp_kernel<<<N_pad / 64, 128, 0, stream>>>(
            agg, w1t, w2t, b1 + (size_t)l * 256, b2 + (size_t)l * 128,
            tmp, colsum, colsumsq, N);
        // BN finalize (+ ReLU except last layer)
        bn_finalize_kernel<<<(N * D_EMB + 255) / 256, 256, 0, stream>>>(
            tmp, colsum, colsumsq,
            gamma + (size_t)l * D_EMB, beta + (size_t)l * D_EMB,
            h, N, (l < 4) ? 1 : 0);
    }
}
